// GRU_15461882266204
// MI455X (gfx1250) — compile-verified, matrix operand order checked by reading, not run on hardware
//
#include <hip/hip_runtime.h>
#include <hip/hip_bf16.h>

#define DIM  2048
#define HID  5632
#define SEQ  8192
#define CONV 4
#define BK   64

typedef __attribute__((ext_vector_type(16))) __bf16 v16bf;
typedef __attribute__((ext_vector_type(8)))  float  v8f;

union FragAB {
    v16bf v;
    uint4 q[2];
};

// ---------------------------------------------------------------------------
// gfx1250 async global -> LDS (ASYNCcnt-tracked), 16B per lane
// LDS dest operand = low 32 bits of the flat shared pointer (LDS offset).
// ---------------------------------------------------------------------------
__device__ __forceinline__ void async_copy_b128(void* lds_ptr, const void* gptr) {
    unsigned lds_off = (unsigned)(unsigned long long)(uintptr_t)lds_ptr;
    asm volatile("global_load_async_to_lds_b128 %0, %1, off"
                 :: "v"(lds_off), "v"(gptr)
                 : "memory");
}

__device__ __forceinline__ void wait_async_zero() {
#if __has_builtin(__builtin_amdgcn_s_wait_asynccnt)
    __builtin_amdgcn_s_wait_asynccnt(0);
#else
    asm volatile("s_wait_asynccnt 0x0" ::: "memory");
#endif
}

// ---------------------------------------------------------------------------
// fp32 -> bf16 (round to nearest even), flat convert
// ---------------------------------------------------------------------------
__global__ void k_f32_to_bf16(const float* __restrict__ src,
                              unsigned short* __restrict__ dst, int n) {
    int i = blockIdx.x * blockDim.x + threadIdx.x;
    if (i < n) {
        unsigned int u = __float_as_uint(src[i]);
        dst[i] = (unsigned short)((u + 0x7FFFu + ((u >> 16) & 1u)) >> 16);
    }
}

// ---------------------------------------------------------------------------
// C[M,N] = A[M,K] * B[N,K]^T   (A,B bf16 row-major K-contiguous, C fp32)
// block tile 128x128, 8 waves of 64x32, K-step 64, double-buffered LDS
// filled by async global->LDS, 16 bf16 WMMA per wave per K-step
// ---------------------------------------------------------------------------
__global__ __launch_bounds__(256) void k_gemm_bf16(
    const unsigned short* __restrict__ A,
    const unsigned short* __restrict__ B,
    float* __restrict__ C,
    int M, int N, int K)
{
    __shared__ __align__(16) unsigned short As[2][128 * BK];
    __shared__ __align__(16) unsigned short Bs[2][128 * BK];

    const int tid  = threadIdx.x;
    const int lane = tid & 31;
    const int wave = tid >> 5;       // 0..7
    const int wm   = wave & 1;       // M half (0..1)  -> 64 rows
    const int wn   = wave >> 1;      // N quarter (0..3) -> 32 cols
    const int l16  = lane & 15;
    const int g    = lane >> 4;      // 0 or 1: selects K-half chunks

    const long tile_m = (long)blockIdx.x * 128;
    const long tile_n = (long)blockIdx.y * 128;

    // per-thread share of the 128x64 bf16 tile fill: 4 x b128 for A, 4 for B
    auto issue_tile = [&](int buf, int k0) {
#pragma unroll
        for (int i = 0; i < 4; ++i) {
            int j   = tid + i * 256;     // 0..1023
            int row = j >> 3;            // 0..127
            int cc  = (j & 7) * 8;       // ushort offset within row (0..56)
            async_copy_b128(&As[buf][row * BK + cc],
                            &A[(tile_m + row) * (long)K + k0 + cc]);
            async_copy_b128(&Bs[buf][row * BK + cc],
                            &B[(tile_n + row) * (long)K + k0 + cc]);
        }
    };

    v8f acc[4][2] = {};

    issue_tile(0, 0);
    wait_async_zero();
    __syncthreads();

    int p = 0;
    for (int k0 = 0; k0 < K; k0 += BK) {
        if (k0 + BK < K)
            issue_tile(p ^ 1, k0 + BK);   // fill back buffer while computing

        // ---- compute on front buffer: 2 WMMA K-steps of 32
#pragma unroll
        for (int kk = 0; kk < BK; kk += 32) {
            // ISA 16-bit A/B layout: lane l -> row/col l%16,
            // K chunks at {kk+g*8 .. +7} and {kk+16+g*8 .. +7}
            FragAB fa[4], fb[2];
#pragma unroll
            for (int mi = 0; mi < 4; ++mi) {
                int r = wm * 64 + mi * 16 + l16;
                fa[mi].q[0] = *(const uint4*)&As[p][r * BK + kk + g * 8];
                fa[mi].q[1] = *(const uint4*)&As[p][r * BK + kk + 16 + g * 8];
            }
#pragma unroll
            for (int ni = 0; ni < 2; ++ni) {
                int r = wn * 32 + ni * 16 + l16;
                fb[ni].q[0] = *(const uint4*)&Bs[p][r * BK + kk + g * 8];
                fb[ni].q[1] = *(const uint4*)&Bs[p][r * BK + kk + 16 + g * 8];
            }
#pragma unroll
            for (int mi = 0; mi < 4; ++mi)
#pragma unroll
                for (int ni = 0; ni < 2; ++ni)
                    acc[mi][ni] = __builtin_amdgcn_wmma_f32_16x16x32_bf16(
                        false, fa[mi].v, false, fb[ni].v,
                        (short)0, acc[mi][ni], false, false);
        }

        wait_async_zero();   // back-buffer fill complete (this wave's share)
        __syncthreads();     // everyone's share complete + front buffer free
        p ^= 1;
    }

    // ---- store C per 16x16 f32 C/D layout: lane -> col l%16, rows g*8+r
#pragma unroll
    for (int mi = 0; mi < 4; ++mi)
#pragma unroll
        for (int ni = 0; ni < 2; ++ni) {
            long n     = tile_n + wn * 32 + ni * 16 + l16;
            long mbase = tile_m + wm * 64 + mi * 16 + g * 8;
#pragma unroll
            for (int r = 0; r < 8; ++r)
                C[(mbase + r) * (long)N + n] = acc[mi][ni][r];
        }
}

// ---------------------------------------------------------------------------
// causal conv(4) + sigmoid gate + segment-reset linear recurrence + silu gate
// proj layout: (SEQ, 3*HID) fp32 = [w0 | z_pre | tilde_h] per row
// one thread per channel, sequential over time (coalesced across channels)
// ---------------------------------------------------------------------------
__global__ __launch_bounds__(256) void k_scan(
    const float* __restrict__ proj,
    const float* __restrict__ conv_w,   // (HID, CONV)
    const int*   __restrict__ cu, int ncu,
    unsigned short* __restrict__ hpost) // (SEQ, HID) bf16
{
    int c = blockIdx.x * blockDim.x + threadIdx.x;
    if (c >= HID) return;

    const float cw0 = conv_w[c * CONV + 0];
    const float cw1 = conv_w[c * CONV + 1];
    const float cw2 = conv_w[c * CONV + 2];
    const float cw3 = conv_w[c * CONV + 3];

    float zm1 = 0.f, zm2 = 0.f, zm3 = 0.f;  // conv window (previous z_pre)
    float h = 0.f;
    int pos = 0;                            // position within current segment
    int j = 0;
    int next_start = (ncu > 1) ? cu[0] : -1;

    const long rs = 3L * HID;
    for (int t = 0; t < SEQ; ++t) {
        const float* row = proj + (long)t * rs;
        float w0 = row[c];
        float zp = row[HID + c];
        float th = row[2 * HID + c];

        bool start = (j < ncu - 1) && (t == next_start);
        if (start) {
            ++j;
            next_start = (j < ncu - 1) ? cu[j] : -1;
            pos = 0;
        }

        float cv = cw3 * zp;                // taps never cross segment start
        if (pos >= 1) cv += cw2 * zm1;
        if (pos >= 2) cv += cw1 * zm2;
        if (pos >= 3) cv += cw0 * zm3;

        float z = 1.0f / (1.0f + __expf(-cv));
        float a = start ? 0.0f : (1.0f - z);   // reset state at seq starts
        float b = z * th;
        h = a * h + b;

        float sw = w0 / (1.0f + __expf(-w0));  // silu(w0)
        float hp = h * sw;

        unsigned int u = __float_as_uint(hp);
        hpost[(long)t * HID + c] =
            (unsigned short)((u + 0x7FFFu + ((u >> 16) & 1u)) >> 16);

        zm3 = zm2; zm2 = zm1; zm1 = zp; ++pos;
    }
}

// ---------------------------------------------------------------------------
extern "C" void kernel_launch(void* const* d_in, const int* in_sizes, int n_in,
                              void* d_out, int out_size, void* d_ws, size_t ws_size,
                              hipStream_t stream) {
    const float* x      = (const float*)d_in[0];
    const int*   cu     = (const int*)  d_in[1];
    const float* w_w    = (const float*)d_in[2];
    const float* wz_w   = (const float*)d_in[3];
    const float* wh_w   = (const float*)d_in[4];
    const float* wo_w   = (const float*)d_in[5];
    const float* conv_w = (const float*)d_in[6];
    float* out = (float*)d_out;

    // ---- workspace carve-out
    char* ws = (char*)d_ws;
    size_t off = 0;
    unsigned short* x_bf  = (unsigned short*)(ws + off); off += (size_t)SEQ * DIM * 2;
    unsigned short* wcat  = (unsigned short*)(ws + off); off += (size_t)3 * HID * DIM * 2;
    unsigned short* wo_bf = (unsigned short*)(ws + off); off += (size_t)DIM * HID * 2;
    float*          proj  = (float*)(ws + off);          off += (size_t)SEQ * 3 * HID * 4;
    unsigned short* hpost = (unsigned short*)(ws + off); off += (size_t)SEQ * HID * 2;

    const int T = 256;
    auto blocks = [](long n, int t) { return (unsigned)((n + t - 1) / t); };

    // ---- precision staging: fp32 -> bf16; concat {w, wz, wh} into (3*HID, DIM)
    k_f32_to_bf16<<<blocks((long)SEQ * DIM, T), T, 0, stream>>>(x, x_bf, SEQ * DIM);
    k_f32_to_bf16<<<blocks((long)HID * DIM, T), T, 0, stream>>>(w_w,  wcat,                         HID * DIM);
    k_f32_to_bf16<<<blocks((long)HID * DIM, T), T, 0, stream>>>(wz_w, wcat + (size_t)HID * DIM,     HID * DIM);
    k_f32_to_bf16<<<blocks((long)HID * DIM, T), T, 0, stream>>>(wh_w, wcat + (size_t)2 * HID * DIM, HID * DIM);
    k_f32_to_bf16<<<blocks((long)DIM * HID, T), T, 0, stream>>>(wo_w, wo_bf, DIM * HID);

    // ---- fused projection GEMM: (8192 x 2048) * (16896 x 2048)^T -> (8192 x 16896)
    k_gemm_bf16<<<dim3(SEQ / 128, (3 * HID) / 128), 256, 0, stream>>>(
        x_bf, wcat, proj, SEQ, 3 * HID, DIM);

    // ---- conv + gates + recurrence + silu, bf16 out
    k_scan<<<HID / 256, 256, 0, stream>>>(proj, conv_w, cu, in_sizes[1], hpost);

    // ---- output GEMM: (8192 x 5632) * (2048 x 5632)^T -> (8192 x 2048) fp32
    k_gemm_bf16<<<dim3(SEQ / 128, DIM / 128), 256, 0, stream>>>(
        hpost, wo_bf, out, SEQ, DIM, HID);
}
